// DotAttentionLayer_82386062672078
// MI455X (gfx1250) — matrix operand order repaired and checked by reading, not compiled, not run on hardware
//
#include <hip/hip_runtime.h>
#include <hip/hip_bf16.h>

typedef __attribute__((ext_vector_type(16))) __bf16 v16bf;
typedef __attribute__((ext_vector_type(8)))  __bf16 v8bf;
typedef __attribute__((ext_vector_type(4)))  __bf16 v4bf;
typedef __attribute__((ext_vector_type(8)))  float  v8f;

#define B_   128
#define Q_   512
#define T_   512
#define D_   256
#define LDK  264                     // padded LDS row stride in bf16 elems
#define KBYTES (T_ * LDK * 2)        // 270336
#define SMEM_BYTES (KBYTES + T_ * 4) // + c[t] array

// Load a 32x16 B-fragment (B[k,n] = K[t0+n, d0+k]) from LDS.
// Lane layout (contiguous k, same order as the A fragments): lane l holds N = l%16, K = e + 16*(l/16), e=0..15 contiguous.
__device__ __forceinline__ v16bf load_bfrag(const __bf16* p) {
  v8bf x0 = *(const v8bf*)p;        // 16B aligned
  v8bf x1 = *(const v8bf*)(p + 8);
  return __builtin_shufflevector(x0, x1, 0,1,2,3,4,5,6,7,8,9,10,11,12,13,14,15);
}

__device__ __forceinline__ float rowmax16(float x) {
  x = fmaxf(x, __shfl_xor(x, 1)); x = fmaxf(x, __shfl_xor(x, 2));
  x = fmaxf(x, __shfl_xor(x, 4)); x = fmaxf(x, __shfl_xor(x, 8));
  return x;
}
__device__ __forceinline__ float rowsum16(float x) {
  x += __shfl_xor(x, 1); x += __shfl_xor(x, 2);
  x += __shfl_xor(x, 4); x += __shfl_xor(x, 8);
  return x;
}

// S-tile for one 32-wide T block: acc0 = cols t0..t0+15, acc1 = cols t0+16..t0+31
__device__ __forceinline__ void score_block(const v16bf* aQ, const __bf16* Klds,
                                            int t0, int laneLo, int hi,
                                            v8f& acc0, v8f& acc1) {
  const __bf16* bp0 = Klds + (size_t)(t0 + laneLo) * LDK + hi * 16;
  const __bf16* bp1 = bp0 + 16 * LDK;
  v8f a0 = {}, a1 = {};
#pragma unroll
  for (int kk = 0; kk < 8; ++kk) {
    v16bf b0 = load_bfrag(bp0 + kk * 32);
    v16bf b1 = load_bfrag(bp1 + kk * 32);
    a0 = __builtin_amdgcn_wmma_f32_16x16x32_bf16(false, aQ[kk], false, b0,
                                                 (short)0, a0, false, false);
    a1 = __builtin_amdgcn_wmma_f32_16x16x32_bf16(false, aQ[kk], false, b1,
                                                 (short)0, a1, false, false);
  }
  acc0 = a0; acc1 = a1;
}

__global__ __launch_bounds__(256, 1)
void attn_pool_kernel(const float* __restrict__ inputs,
                      const float* __restrict__ query,
                      float* __restrict__ ws) {
  extern __shared__ char smem_raw[];
  __bf16* Klds = (__bf16*)smem_raw;
  float*  cSum = (float*)(smem_raw + KBYTES);

  const int wg   = blockIdx.x;
  const int b    = wg >> 2;   // batch
  const int qt   = wg & 3;    // query tile (128 rows each)
  const int tid  = threadIdx.x;
  const int lane = tid & 31;
  const int wave = tid >> 5;
  const int laneLo = lane & 15;
  const int hi     = lane >> 4;

  // ---- Stage K (== V) for this batch into LDS as bf16 (padded rows) ----
  const float* Kg = inputs + (size_t)b * T_ * D_;
  for (int i = tid; i < T_ * (D_ / 4); i += 256) {
    const int row = i >> 6;        // D_/4 == 64
    const int c4  = i & 63;
    const float4 f = ((const float4*)Kg)[i];
    v4bf h;
    h[0] = (__bf16)f.x; h[1] = (__bf16)f.y; h[2] = (__bf16)f.z; h[3] = (__bf16)f.w;
    *(v4bf*)(Klds + (size_t)row * LDK + c4 * 4) = h;
  }
  // zero weight-column-sum array (used in pass 2)
  for (int i = tid; i < T_; i += 256) cSum[i] = 0.0f;
  __syncthreads();

  // ---- Load Q A-fragments once into registers (contiguous-k A 16x32 layout, matching B) ----
  // lane holds row q0 + l%16; elem e -> d = kk*32 + e + hi*16
  const int q0 = qt * 128 + wave * 16;
  const float* qrow = query + ((size_t)b * Q_ + (size_t)(q0 + laneLo)) * D_;
  v16bf aQ[8];
#pragma unroll
  for (int kk = 0; kk < 8; ++kk) {
    const int dbase = kk * 32 + hi * 16;
    const float4 f0 = *(const float4*)(qrow + dbase);
    const float4 f1 = *(const float4*)(qrow + dbase + 4);
    const float4 f2 = *(const float4*)(qrow + dbase + 8);
    const float4 f3 = *(const float4*)(qrow + dbase + 12);
    v16bf a;
    a[0]=(__bf16)f0.x;  a[1]=(__bf16)f0.y;  a[2]=(__bf16)f0.z;  a[3]=(__bf16)f0.w;
    a[4]=(__bf16)f1.x;  a[5]=(__bf16)f1.y;  a[6]=(__bf16)f1.z;  a[7]=(__bf16)f1.w;
    a[8]=(__bf16)f2.x;  a[9]=(__bf16)f2.y;  a[10]=(__bf16)f2.z; a[11]=(__bf16)f2.w;
    a[12]=(__bf16)f3.x; a[13]=(__bf16)f3.y; a[14]=(__bf16)f3.z; a[15]=(__bf16)f3.w;
    aQ[kk] = a;
  }

  const float scale = 0.0625f;  // 1/sqrt(256)

  // Per-lane row state: VGPR slot v <-> query row q0 + v + 8*hi (C-layout)
  float m[8], l[8];
#pragma unroll
  for (int v = 0; v < 8; ++v) { m[v] = -1e30f; l[v] = 0.0f; }

  // ---- PASS 1: online row max + row sum of exp ----
  for (int t0 = 0; t0 < T_; t0 += 32) {
    v8f acc0, acc1;
    score_block(aQ, Klds, t0, laneLo, hi, acc0, acc1);
#pragma unroll
    for (int v = 0; v < 8; ++v) {
      const float s0 = acc0[v] * scale;
      const float s1 = acc1[v] * scale;
      const float bm = rowmax16(fmaxf(s0, s1));
      const float mn = fmaxf(m[v], bm);
      const float rs = rowsum16(__expf(s0 - mn) + __expf(s1 - mn));
      l[v] = l[v] * __expf(m[v] - mn) + rs;
      m[v] = mn;
    }
  }

  float rl[8];
#pragma unroll
  for (int v = 0; v < 8; ++v) rl[v] = 1.0f / l[v];

  // ---- PASS 2: recompute scores, accumulate weight column sums c[t] ----
  for (int t0 = 0; t0 < T_; t0 += 32) {
    v8f acc0, acc1;
    score_block(aQ, Klds, t0, laneLo, hi, acc0, acc1);
    float cs0 = 0.0f, cs1 = 0.0f;
#pragma unroll
    for (int v = 0; v < 8; ++v) {
      cs0 += __expf(acc0[v] * scale - m[v]) * rl[v];
      cs1 += __expf(acc1[v] * scale - m[v]) * rl[v];
    }
    // column j lives at lane l%16 in both halves: fold halves, one atomic per col
    cs0 += __shfl_xor(cs0, 16);
    cs1 += __shfl_xor(cs1, 16);
    if (lane < 16) {
      atomicAdd(&cSum[t0 + lane], cs0);
      atomicAdd(&cSum[t0 + 16 + lane], cs1);
    }
  }
  __syncthreads();

  // ---- GEMV: partial out[b,:] = c^T K ; thread tid owns column d = tid ----
  float y = 0.0f;
  for (int t = 0; t < T_; t += 4) {
    const float4 cv = *(const float4*)&cSum[t];
    y += cv.x * (float)Klds[(size_t)(t + 0) * LDK + tid];
    y += cv.y * (float)Klds[(size_t)(t + 1) * LDK + tid];
    y += cv.z * (float)Klds[(size_t)(t + 2) * LDK + tid];
    y += cv.w * (float)Klds[(size_t)(t + 3) * LDK + tid];
  }
  ws[((size_t)qt * B_ + b) * D_ + tid] = y;
}

__global__ __launch_bounds__(256, 1)
void reduce_kernel(const float* __restrict__ ws, float* __restrict__ out) {
  const int i = blockIdx.x * 256 + threadIdx.x;  // over B_*D_ = 32768
  const int n = B_ * D_;
  out[i] = ws[i] + ws[n + i] + ws[2 * n + i] + ws[3 * n + i];
}

extern "C" void kernel_launch(void* const* d_in, const int* in_sizes, int n_in,
                              void* d_out, int out_size, void* d_ws, size_t ws_size,
                              hipStream_t stream) {
  const float* inputs = (const float*)d_in[0];  // [B, T, D] fp32
  const float* query  = (const float*)d_in[1];  // [B, Q, D] fp32
  float* ws  = (float*)d_ws;                    // 4 * B * D partials
  float* out = (float*)d_out;                   // [B, D] fp32

  attn_pool_kernel<<<dim3(B_ * 4), dim3(256), SMEM_BYTES, stream>>>(inputs, query, ws);
  reduce_kernel<<<dim3(B_ * D_ / 256), dim3(256), 0, stream>>>(ws, out);
}
